// MultiHeadAttention_21577915695182
// MI455X (gfx1250) — matrix operand order repaired and checked by reading, not compiled, not run on hardware
//
#include <hip/hip_runtime.h>

typedef __bf16 bf16_t;
typedef __attribute__((ext_vector_type(16))) __bf16 v16bf;
typedef __attribute__((ext_vector_type(2)))  __bf16 v2bf;
typedef __attribute__((ext_vector_type(8)))  float  v8f;
typedef __attribute__((ext_vector_type(4)))  float  f32x4;
typedef __attribute__((ext_vector_type(4)))  int    i32x4;
typedef __attribute__((ext_vector_type(8)))  int    i32x8;
typedef __attribute__((ext_vector_type(4)))  unsigned int u32x4;
typedef __attribute__((ext_vector_type(2)))  unsigned int u32x2;

#define LOG2E 1.44269504088896340736f

#if defined(__has_builtin)
#if __has_builtin(__builtin_amdgcn_cvt_pk_bf16_f32)
#define HAVE_CVT_PK_BF16 1
#endif
#if __has_builtin(__builtin_amdgcn_permlane16)
#define HAVE_PERMLANE16 1
#endif
#endif

__device__ __forceinline__ unsigned f2bfu_sw(float f) {
  union { float f; unsigned u; } a; a.f = f;
  unsigned r = a.u + 0x7FFFu + ((a.u >> 16) & 1u);  // round-to-nearest-even
  return r >> 16;
}
// pack two f32 -> two bf16 in one dword
__device__ __forceinline__ unsigned pk2bf(float lo, float hi) {
#ifdef HAVE_CVT_PK_BF16
  union { v2bf v; unsigned u; } u;
  u.v = __builtin_amdgcn_cvt_pk_bf16_f32(lo, hi);
  return u.u;
#else
  return f2bfu_sw(lo) | (f2bfu_sw(hi) << 16);
#endif
}
__device__ __forceinline__ bf16_t f2bf(float f) {
#ifdef HAVE_CVT_PK_BF16
  v2bf v = __builtin_amdgcn_cvt_pk_bf16_f32(f, f);
  return v.x;
#else
  union { unsigned short s; bf16_t b; } o; o.s = (unsigned short)f2bfu_sw(f);
  return o.b;
#endif
}

// max-reduce across xor masks 1,2,4,8 (within 16-lane halves)
__device__ __forceinline__ float xmax16(float t) {
#ifdef HAVE_PERMLANE16
  union { float f; unsigned u; } a, b;
  // select tables: lane i reads lane i^m within its 16-lane row
  a.f = t; b.u = __builtin_amdgcn_permlane16(a.u, a.u, 0x67452301u, 0xEFCDAB89u, false, false);
  t = fmaxf(t, b.f);
  a.f = t; b.u = __builtin_amdgcn_permlane16(a.u, a.u, 0x54761032u, 0xDCFE98BAu, false, false);
  t = fmaxf(t, b.f);
  a.f = t; b.u = __builtin_amdgcn_permlane16(a.u, a.u, 0x32107654u, 0xBA98FEDCu, false, false);
  t = fmaxf(t, b.f);
  a.f = t; b.u = __builtin_amdgcn_permlane16(a.u, a.u, 0xFEDCBA98u, 0x76543210u, false, false);
  t = fmaxf(t, b.f);
#else
  t = fmaxf(t, __shfl_xor(t, 1, 32));
  t = fmaxf(t, __shfl_xor(t, 2, 32));
  t = fmaxf(t, __shfl_xor(t, 4, 32));
  t = fmaxf(t, __shfl_xor(t, 8, 32));
#endif
  return t;
}

__device__ __forceinline__ v8f wmma_bf16(v16bf a, v16bf b, v8f c) {
  return __builtin_amdgcn_wmma_f32_16x16x32_bf16(false, a, false, b, (short)0, c,
                                                 false, false);
}

__device__ __forceinline__ v16bf ones_frag() {
  union { v16bf v; unsigned u[8]; } t;
#pragma unroll
  for (int i = 0; i < 8; ++i) t.u[i] = 0x3F803F80u;  // bf16 1.0 pair
  return t.v;
}

__device__ __forceinline__ v16bf load_frag_2x16B(const bf16_t* p0, const bf16_t* p1) {
  union { v16bf v; i32x4 q[2]; } u;
  u.q[0] = *(const i32x4*)p0;
  u.q[1] = *(const i32x4*)p1;
  return u.v;
}
// A fragment (16x32 bf16, M in lanes)
__device__ __forceinline__ v16bf load_a_frag(const bf16_t* rowp, int kstart, int lane) {
  int klo = (lane >> 4) << 3;
  return load_frag_2x16B(rowp + kstart + klo, rowp + kstart + 16 + klo);
}
// B fragment (32x16, N in lanes) from N-major [n][k] tile
__device__ __forceinline__ v16bf load_b_frag(const bf16_t* rowp, int kstart, int lane) {
  const bf16_t* p = rowp + kstart + ((lane >> 4) << 3);
  return load_frag_2x16B(p, p + 16);
}

// ---------------------------------------------------------------------------
// TDM: 2D bf16 tile global -> LDS with per-row LDS padding.
// pad codes: interval 2^(v+1) DWORDs, amount (v+1) DWORDs.
// ---------------------------------------------------------------------------
__device__ __forceinline__ void tdm_load_tile_2d_bf16(
    unsigned lds_off, const void* gptr, unsigned tile_d, unsigned tile_rows,
    unsigned tensor_rows, unsigned row_stride_elems,
    unsigned pad_interval_code, unsigned pad_amount_code) {
  unsigned long long ga = (unsigned long long)(uintptr_t)gptr;
  u32x4 g0;
  g0.x = 1u;                                            // count=1, user mode
  g0.y = lds_off;                                       // lds_addr
  g0.z = (unsigned)(ga & 0xFFFFFFFFu);                  // global_addr[31:0]
  g0.w = (unsigned)((ga >> 32) & 0x01FFFFFFu) | 0x80000000u;  // addr[56:32] | type=2
  i32x8 g1;
  g1[0] = (int)((1u << 16) | (1u << 20) | (pad_interval_code << 22) | (pad_amount_code << 25));
  g1[1] = (int)((tile_d & 0xFFFFu) << 16);              // tensor_dim0[15:0]
  g1[2] = (int)(((tile_d >> 16) & 0xFFFFu) | ((tensor_rows & 0xFFFFu) << 16));
  g1[3] = (int)(((tensor_rows >> 16) & 0xFFFFu) | ((tile_d & 0xFFFFu) << 16)); // tile_dim0
  g1[4] = (int)(tile_rows & 0xFFFFu);                   // tile_dim1, tile_dim2=0
  g1[5] = (int)row_stride_elems;                        // tensor_dim0_stride[31:0]
  g1[6] = 0;
  g1[7] = 0;
  i32x4 z4 = {0, 0, 0, 0};
  i32x8 z8 = {0, 0, 0, 0, 0, 0, 0, 0};
  __builtin_amdgcn_tensor_load_to_lds(g0, g1, z4, z4, z8, 0);
}

// ---------------------------------------------------------------------------
// Tiled GEMM: out[M][Nn] = A[M][K] * Bw[Nn][K]^T (+bias). 8 waves, 4(M)x2(N),
// block tile 128x64, K-step 32. bf16 A-tile comes in via TDM.
// ---------------------------------------------------------------------------
constexpr int ASTR = 40;  // 80B LDS row stride

struct alignas(16) GemmSmem {
  bf16_t a[128 * ASTR];
  bf16_t b[64 * ASTR];
};

template <bool A_IS_F32, bool BIAS, bool OUT_F32>
__global__ __launch_bounds__(256) void gemm_bf16_kernel(
    const void* __restrict__ Aptr, const float* __restrict__ Bw,
    const float* __restrict__ bias, void* __restrict__ Optr,
    int K, int ldo) {
  __shared__ GemmSmem sm;
  const int tid = threadIdx.x, lane = tid & 31, w = tid >> 5;
  const int mw = w & 3, nw = w >> 2;
  const int blockM = blockIdx.y * 128;
  const int blockN = blockIdx.x * 64;
  const float*  Af = (const float*)Aptr;
  const bf16_t* Ab = (const bf16_t*)Aptr;
  const unsigned a_lds_off = (unsigned)(uintptr_t)&sm.a[0];

  const v8f vzero = {0.f, 0.f, 0.f, 0.f, 0.f, 0.f, 0.f, 0.f};
  v8f acc[2][2];
#pragma unroll
  for (int i = 0; i < 2; ++i)
#pragma unroll
    for (int j = 0; j < 2; ++j) acc[i][j] = vzero;

  for (int kk = 0; kk < K; kk += 32) {
    __syncthreads();
    if constexpr (A_IS_F32) {
#pragma unroll
      for (int i = 0; i < 4; ++i) {           // A tile 128x32 f32, float4 loads
        int idx = i * 256 + tid;
        int r = idx >> 3, c = (idx & 7) * 4;
        f32x4 v = *(const f32x4*)&Af[(size_t)(blockM + r) * K + kk + c];
        u32x2 pk = {pk2bf(v.x, v.y), pk2bf(v.z, v.w)};
        *(u32x2*)&sm.a[r * ASTR + c] = pk;
      }
    } else {
      // A tile 128x32 bf16 via TDM: rows 64B, pad 16B -> 80B stride (codes 3,3)
      if (w == 0)
        tdm_load_tile_2d_bf16(a_lds_off, &Ab[(size_t)blockM * K + kk],
                              32u, 128u, 8192u, (unsigned)K, 3u, 3u);
    }
#pragma unroll
    for (int i = 0; i < 2; ++i) {             // B tile 64x32 f32 (N-major), float4
      int idx = i * 256 + tid;
      int r = idx >> 3, c = (idx & 7) * 4;
      f32x4 v = *(const f32x4*)&Bw[(size_t)(blockN + r) * K + kk + c];
      u32x2 pk = {pk2bf(v.x, v.y), pk2bf(v.z, v.w)};
      *(u32x2*)&sm.b[r * ASTR + c] = pk;
    }
    if (kk + 32 < K && tid < 192) {           // prefetch next K-slab
      if (tid < 128) {
        const void* p = A_IS_F32 ? (const void*)&Af[(size_t)(blockM + tid) * K + kk + 32]
                                 : (const void*)&Ab[(size_t)(blockM + tid) * K + kk + 32];
        __builtin_prefetch(p, 0, 1);
      } else {
        __builtin_prefetch(&Bw[(size_t)(blockN + (tid & 63)) * K + kk + 32], 0, 1);
      }
    }
    if constexpr (!A_IS_F32) {
      if (w == 0) __builtin_amdgcn_s_wait_tensorcnt(0);
    }
    __syncthreads();
    v16bf afr[2], bfr[2];
#pragma unroll
    for (int i = 0; i < 2; ++i)
      afr[i] = load_a_frag(&sm.a[(mw * 32 + i * 16 + (lane & 15)) * ASTR], 0, lane);
#pragma unroll
    for (int j = 0; j < 2; ++j)
      bfr[j] = load_b_frag(&sm.b[(nw * 32 + j * 16 + (lane & 15)) * ASTR], 0, lane);
#pragma unroll
    for (int i = 0; i < 2; ++i)
#pragma unroll
      for (int j = 0; j < 2; ++j) acc[i][j] = wmma_bf16(afr[i], bfr[j], acc[i][j]);
  }

  const int rbase = (lane >> 4) << 3;
  const int col = lane & 15;
#pragma unroll
  for (int i = 0; i < 2; ++i)
#pragma unroll
    for (int j = 0; j < 2; ++j) {
      int n = blockN + nw * 32 + j * 16 + col;
      float bv = BIAS ? bias[n] : 0.f;
#pragma unroll
      for (int r = 0; r < 8; ++r) {
        int m = blockM + mw * 32 + i * 16 + rbase + r;
        float v = acc[i][j][r] + bv;
        if constexpr (OUT_F32) ((float*)Optr)[(size_t)m * ldo + n] = v;
        else                   ((bf16_t*)Optr)[(size_t)m * ldo + n] = f2bf(v);
      }
    }
}

// ---------------------------------------------------------------------------
// Flash attention. Grid (16 qblocks, H=16, B=4), 256 thr = 8 waves.
// Wave: 16 query rows. 64-key K/V tiles streamed via LDS; K tile via TDM.
// Row-sums of P accumulated with a ones-matrix WMMA (no shuffle reduction).
// ---------------------------------------------------------------------------
constexpr int QSTR = 72;  // 144B rows (Q stage / P tiles)
constexpr int KSTR = 72;  // 144B rows (K tile [key][d]) == TDM padded stride
constexpr int VSTR = 80;  // 160B rows (V^T tile [d][key])

struct alignas(16) AttnSmem {
  union {
    bf16_t q[128 * QSTR];
    bf16_t p[128 * QSTR];  // wave w owns rows w*16..w*16+15
  };
  bf16_t k[64 * KSTR];
  bf16_t vt[64 * VSTR];
};

__global__ __launch_bounds__(256) void flash_attn_kernel(
    const bf16_t* __restrict__ qkv, bf16_t* __restrict__ attn_out) {
  __shared__ AttnSmem sm;
  const int tid = threadIdx.x, lane = tid & 31, w = tid >> 5;
  const int qb = blockIdx.x, h = blockIdx.y, b = blockIdx.z;
  const int N = 2048, C3 = 3072;
  const size_t base = (size_t)b * N * C3 + (size_t)h * 64;
  const bf16_t* Qg = qkv + base;
  const bf16_t* Kg = qkv + base + 1024;
  const bf16_t* Vg = qkv + base + 2048;
  const int q0 = qb * 128;
  const float SCALE = 0.125f;  // 64^-0.5

#pragma unroll
  for (int i = 0; i < 4; ++i) {  // stage Q 128x64 via b128
    int idx = i * 256 + tid;
    int r = idx >> 3, c = (idx & 7) * 8;
    *(i32x4*)&sm.q[r * QSTR + c] = *(const i32x4*)&Qg[(size_t)(q0 + r) * C3 + c];
  }
  __syncthreads();
  v16bf aq[2];
  {
    const bf16_t* rowp = &sm.q[(w * 16 + (lane & 15)) * QSTR];
    aq[0] = load_a_frag(rowp, 0, lane);
    aq[1] = load_a_frag(rowp, 32, lane);
  }
  __syncthreads();  // Q reads done before P overwrites the union

  const v8f vzero = {0.f, 0.f, 0.f, 0.f, 0.f, 0.f, 0.f, 0.f};
  const v16bf vones = ones_frag();
  v8f o[4], lacc = vzero;
#pragma unroll
  for (int j = 0; j < 4; ++j) o[j] = vzero;
  float mrun[8];
#pragma unroll
  for (int r = 0; r < 8; ++r) mrun[r] = -3.0e38f;

  bf16_t* pw = &sm.p[(w * 16) * QSTR];
  const int rbase = (lane >> 4) << 3;
  const int col = lane & 15;
  const unsigned k_lds_off = (unsigned)(uintptr_t)&sm.k[0];

  for (int kt = 0; kt < 32; ++kt) {
    const int k0 = kt * 64;
    if (w == 0) {  // K tile [64 keys][64 d] via TDM (rows 128B, pad 16B -> 144B)
      tdm_load_tile_2d_bf16(k_lds_off, &Kg[(size_t)k0 * C3], 64u, 64u, 2048u,
                            (unsigned)C3, 4u, 3u);
    }
#pragma unroll
    for (int i = 0; i < 2; ++i) {  // V tile, transposed into vt[d][key]
      int idx = i * 256 + tid;
      int r = idx >> 3, c = (idx & 7) * 8;
      union { i32x4 q; bf16_t e[8]; } u;
      u.q = *(const i32x4*)&Vg[(size_t)(k0 + r) * C3 + c];
#pragma unroll
      for (int e = 0; e < 8; ++e) sm.vt[(c + e) * VSTR + r] = u.e[e];
    }
    if (kt + 1 < 32 && tid < 128) {  // prefetch next key tile
      int row = k0 + 64 + (tid & 63);
      const bf16_t* pf = (tid < 64) ? &Kg[(size_t)row * C3] : &Vg[(size_t)row * C3];
      __builtin_prefetch(pf, 0, 1);
    }
    if (w == 0) __builtin_amdgcn_s_wait_tensorcnt(0);
    __syncthreads();

    // S = (Q K^T) * scale
    v16bf kf[4];
#pragma unroll
    for (int j = 0; j < 4; ++j)
      kf[j] = load_b_frag(&sm.k[(j * 16 + (lane & 15)) * KSTR], 0, lane);
    v8f s[4];
#pragma unroll
    for (int j = 0; j < 4; ++j) s[j] = wmma_bf16(aq[0], kf[j], vzero);
#pragma unroll
    for (int j = 0; j < 4; ++j)
      kf[j] = load_b_frag(&sm.k[(j * 16 + (lane & 15)) * KSTR], 32, lane);
#pragma unroll
    for (int j = 0; j < 4; ++j) s[j] = wmma_bf16(aq[1], kf[j], s[j]);
#pragma unroll
    for (int j = 0; j < 4; ++j) s[j] *= SCALE;

    // online softmax: max via permlane16; alpha-rescale O and row-sum acc
#pragma unroll
    for (int r = 0; r < 8; ++r) {
      float t = xmax16(fmaxf(fmaxf(s[0][r], s[1][r]), fmaxf(s[2][r], s[3][r])));
      float mn = fmaxf(mrun[r], t);
      float alpha = exp2f((mrun[r] - mn) * LOG2E);
      mrun[r] = mn;
#pragma unroll
      for (int j = 0; j < 4; ++j) s[j][r] = exp2f((s[j][r] - mn) * LOG2E);
      lacc[r] *= alpha;
#pragma unroll
      for (int j = 0; j < 4; ++j) o[j][r] *= alpha;
    }

    // P -> LDS (same-wave LDS is in-order); convert row pairs with one cvt_pk
#pragma unroll
    for (int j = 0; j < 4; ++j)
#pragma unroll
      for (int r = 0; r < 8; r += 2) {
        unsigned pk = pk2bf(s[j][r], s[j][r + 1]);
        union { unsigned short s[2]; unsigned u; } uu; uu.u = pk;
        union { unsigned short s; bf16_t b; } b0, b1;
        b0.s = uu.s[0]; b1.s = uu.s[1];
        pw[(rbase + r) * QSTR + j * 16 + col] = b0.b;
        pw[(rbase + r + 1) * QSTR + j * 16 + col] = b1.b;
      }
    __syncthreads();

    // O += P @ V ; lacc += P @ ones (row sums)
#pragma unroll
    for (int ks = 0; ks < 2; ++ks) {
      v16bf ap = load_a_frag(&pw[(lane & 15) * QSTR], ks * 32, lane);
      v16bf vf[4];
#pragma unroll
      for (int j = 0; j < 4; ++j)
        vf[j] = load_b_frag(&sm.vt[(j * 16 + (lane & 15)) * VSTR], ks * 32, lane);
#pragma unroll
      for (int j = 0; j < 4; ++j) o[j] = wmma_bf16(ap, vf[j], o[j]);
      lacc = wmma_bf16(ap, vones, lacc);
    }
    __syncthreads();
  }

  // normalize + store [B,N,H*D] bf16 for the projection GEMM
  const size_t obase = (size_t)b * N * 1024 + (size_t)h * 64;
#pragma unroll
  for (int r = 0; r < 8; ++r) {
    float inv = 1.0f / lacc[r];
    int n = q0 + w * 16 + rbase + r;
#pragma unroll
    for (int j = 0; j < 4; ++j)
      attn_out[obase + (size_t)n * 1024 + j * 16 + col] = f2bf(o[j][r] * inv);
  }
}

extern "C" void kernel_launch(void* const* d_in, const int* in_sizes, int n_in,
                              void* d_out, int out_size, void* d_ws, size_t ws_size,
                              hipStream_t stream) {
  (void)in_sizes; (void)n_in; (void)out_size; (void)ws_size;
  const float* x      = (const float*)d_in[0];
  const float* w_qkv  = (const float*)d_in[1];
  const float* w_proj = (const float*)d_in[2];
  const float* b_proj = (const float*)d_in[3];

  bf16_t* qkv  = (bf16_t*)d_ws;                                     // 48 MB
  bf16_t* attn = (bf16_t*)((char*)d_ws + (size_t)8192 * 3072 * 2);  // 16 MB

  gemm_bf16_kernel<true, false, false>
      <<<dim3(48, 64), 256, 0, stream>>>(x, w_qkv, nullptr, qkv, 1024, 3072);
  flash_attn_kernel<<<dim3(16, 16, 4), 256, 0, stream>>>(qkv, attn);
  gemm_bf16_kernel<false, true, true>
      <<<dim3(16, 64), 256, 0, stream>>>(attn, w_proj, b_proj, d_out, 1024, 1024);
}
